// Attention_54778012893268
// MI455X (gfx1250) — matrix operand order repaired and checked by reading, not compiled, not run on hardware
//
#include <hip/hip_runtime.h>

typedef __attribute__((ext_vector_type(16))) __bf16 v16bf;
typedef __attribute__((ext_vector_type(8)))  float  v8f;

typedef unsigned short u16;
typedef unsigned int   u32;

// ---------- bf16 helpers (bit-level, round-to-nearest-even) ----------
__device__ __forceinline__ u16 f2bf(float f) {
    u32 u = __float_as_uint(f);
    u32 r = u + 0x7FFFu + ((u >> 16) & 1u);
    return (u16)(r >> 16);
}
__device__ __forceinline__ float bf2f(u16 h) {
    return __uint_as_float(((u32)h) << 16);
}

// ---------- CDNA5 async global->LDS copy (ASYNCcnt-tracked, no VGPR bounce) ----------
// LDS address is taken from the vdst VGPR; low 32 bits of a generic pointer to __shared__
// are exactly the wave-relative LDS byte offset (ISA 10.2 aperture mapping).
__device__ __forceinline__ void async_b128(void* lds, const void* gaddr) {
    unsigned loff = (unsigned)(size_t)lds;
    asm volatile("global_load_async_to_lds_b128 %0, %1, off"
                 :: "v"(loff), "v"(gaddr) : "memory");
}
__device__ __forceinline__ void wait_async0() {
    asm volatile("s_wait_asynccnt 0x0" ::: "memory");
}

union FragBits { u32 u[8]; v16bf v; };

// ---------- WMMA fragment loaders (contiguous k order, A and B alike, wave32) ----------
// A matrix 16x32 bf16: lane m = lane&15, g = lane>>4.
// VGPR v holds K = g*16 + v*2, +1   (packed pair)
__device__ __forceinline__ v16bf load_frag_a(const u16* lds, int row0, int ld, int k0) {
    int lane = threadIdx.x & 31;
    int m = lane & 15, g = lane >> 4;
    FragBits f;
#pragma unroll
    for (int v = 0; v < 8; ++v) {
        int k = g * 16 + v * 2;
        f.u[v] = *(const u32*)(lds + (size_t)(row0 + m) * ld + k0 + k);
    }
    return f.v;
}

// B matrix 32x16 bf16 from an N-major (row = output column n, col = k) tile:
// lane: n = lane&15, kbase = (lane>>4)*16; VGPR v holds K = kbase+2v, +1
__device__ __forceinline__ v16bf load_frag_b_nt(const u16* lds, int n0, int ld, int k0) {
    int lane = threadIdx.x & 31;
    int n = lane & 15, kb = (lane >> 4) * 16;
    FragBits f;
#pragma unroll
    for (int v = 0; v < 8; ++v) {
        f.u[v] = *(const u32*)(lds + (size_t)(n0 + n) * ld + k0 + kb + v * 2);
    }
    return f.v;
}

// B matrix 32x16 bf16 from a K-major (row = k, col = n) tile (used for V)
__device__ __forceinline__ v16bf load_frag_b_kn(const u16* lds, int n0, int ld, int k0) {
    int lane = threadIdx.x & 31;
    int n = lane & 15, kb = (lane >> 4) * 16;
    FragBits f;
#pragma unroll
    for (int v = 0; v < 8; ++v) {
        int k = k0 + kb + v * 2;
        u32 lo = lds[(size_t)k * ld + n0 + n];
        u32 hi = lds[(size_t)(k + 1) * ld + n0 + n];
        f.u[v] = lo | (hi << 16);
    }
    return f.v;
}

__device__ __forceinline__ v8f wmma_bf16(v16bf a, v16bf b, v8f c) {
    return __builtin_amdgcn_wmma_f32_16x16x32_bf16(false, a, false, b, (short)0, c, false, false);
}

__device__ __forceinline__ v8f zero8() {
    v8f z = {0.f, 0.f, 0.f, 0.f, 0.f, 0.f, 0.f, 0.f};
    return z;
}

// ---------- kernel 1: fp32 -> bf16 cast ----------
__global__ __launch_bounds__(256) void cast_bf16_kernel(const float* __restrict__ src,
                                                        u16* __restrict__ dst, int n) {
    int i = blockIdx.x * 256 + threadIdx.x;
    if (i < n) dst[i] = f2bf(src[i]);
}

// ---------- kernel 2: GEMM  C[M,N] = A[M,K] * Bt[N,K]^T  (bf16 in, f32 acc) ----------
__device__ __forceinline__ void store_out(float* p, float v) { *p = v; }
__device__ __forceinline__ void store_out(u16* p, float v)   { *p = f2bf(v); }

template <typename OutT>
__global__ __launch_bounds__(256) void gemm_bf16_nt(const u16* __restrict__ A,
                                                    const u16* __restrict__ Bt,
                                                    OutT* __restrict__ C,
                                                    int M, int N, int K) {
    constexpr int LD = 40;                 // 32 + 8 pad: conflict-free; 80B row = 16B-aligned
    __shared__ u16 As[2][128 * LD];
    __shared__ u16 Bs[2][128 * LD];
    const int tid  = threadIdx.x;
    const int lane = tid & 31, wave = tid >> 5;
    const int tileM = blockIdx.y * 128, tileN = blockIdx.x * 128;
    const int wm = (wave >> 2) * 64, wn = (wave & 3) * 32; // 2x4 wave grid, 64x32 per wave
    const int srow = tid >> 1, shalf = tid & 1;            // stage: 128 rows x 32 halfwords

    const u16* gA = A + (size_t)(tileM + srow) * K + shalf * 16;
    const u16* gB = Bt + (size_t)(tileN + srow) * K + shalf * 16;

    auto stage = [&](int buf, int k0) {
        u16* da = &As[buf][srow * LD + shalf * 16];
        u16* db = &Bs[buf][srow * LD + shalf * 16];
        async_b128(da,     gA + k0);
        async_b128(da + 8, gA + k0 + 8);
        async_b128(db,     gB + k0);
        async_b128(db + 8, gB + k0 + 8);
    };

    v8f acc[4][2];
#pragma unroll
    for (int i = 0; i < 4; ++i)
#pragma unroll
        for (int j = 0; j < 2; ++j) acc[i][j] = zero8();

    stage(0, 0);
    wait_async0();
    __syncthreads();

    int buf = 0;
    for (int k0 = 0; k0 < K; k0 += 32) {
        if (k0 + 32 < K) stage(buf ^ 1, k0 + 32);   // prefetch next tile (async)

        v16bf a[4], b[2];
#pragma unroll
        for (int mf = 0; mf < 4; ++mf) a[mf] = load_frag_a(As[buf], wm + mf * 16, LD, 0);
#pragma unroll
        for (int nf = 0; nf < 2; ++nf) b[nf] = load_frag_b_nt(Bs[buf], wn + nf * 16, LD, 0);
#pragma unroll
        for (int mf = 0; mf < 4; ++mf)
#pragma unroll
            for (int nf = 0; nf < 2; ++nf)
                acc[mf][nf] = wmma_bf16(a[mf], b[nf], acc[mf][nf]);

        wait_async0();      // prefetched tile landed (this wave's copies)
        __syncthreads();    // everyone done reading buf + everyone's copies visible
        buf ^= 1;
    }

    // C/D layout: col = lane&15, row = r + 8*(lane>>4)
    const int g = lane >> 4, n = lane & 15;
#pragma unroll
    for (int mf = 0; mf < 4; ++mf)
#pragma unroll
        for (int nf = 0; nf < 2; ++nf)
#pragma unroll
            for (int r = 0; r < 8; ++r) {
                int rr = tileM + wm + mf * 16 + g * 8 + r;
                int cc = tileN + wn + nf * 16 + n;
                store_out(&C[(size_t)rr * N + cc], acc[mf][nf][r]);
            }
}

// ---------- kernel 3: in-place interleaved RoPE on q,k slices of qkv ----------
// qkv layout: [B*T, 3*H*D] bf16, f = (s*H + h)*D + d, B=4 T=2048 H=16 D=128
__global__ __launch_bounds__(256) void rope_kernel(u16* __restrict__ qkv) {
    u32 i = blockIdx.x * 256u + threadIdx.x;   // 2^24 threads
    int d2 = i & 63;
    int h  = (i >> 6) & 15;
    int s  = (i >> 10) & 1;
    int t  = (i >> 11) & 2047;
    int b  = i >> 22;
    size_t p = (size_t)(b * 2048 + t) * 6144 + (size_t)((s * 16 + h) * 128 + d2 * 2);
    float x1 = bf2f(qkv[p]);
    float x2 = bf2f(qkv[p + 1]);
    float freq = __expf(-((float)(2 * d2) * (1.0f / 128.0f)) * 9.210340371976184f);
    float ang = (float)t * freq;
    float sn, cs;
    __sincosf(ang, &sn, &cs);
    qkv[p]     = f2bf(x1 * cs - x2 * sn);
    qkv[p + 1] = f2bf(x1 * sn + x2 * cs);
}

// ---------- kernel 4: flash attention, 64-row Q tile per workgroup (4 waves) ----------
__global__ __launch_bounds__(128) void flash_attn_kernel(const u16* __restrict__ qkv,
                                                         u16* __restrict__ y) {
    constexpr int LDQ = 136;   // 128 + 8 pad; 272B row = 16B-aligned
    constexpr int LDP = 72;    // 64 + 8 pad
    __shared__ u16 Qs[64 * LDQ];
    __shared__ u16 Ks[2][64 * LDQ];
    __shared__ u16 Vs[2][64 * LDQ];
    __shared__ u16 Ps[64 * LDP];

    const int tid = threadIdx.x, lane = tid & 31, wave = tid >> 5;
    const int qt = blockIdx.x, h = blockIdx.y, b = blockIdx.z;
    const int qbase = qt * 64;
    const int srow = tid >> 1, shalf = tid & 1;     // stage 64 rows x 128 halfwords
    const float scale = 0.08838834764831845f;        // 1/sqrt(128)

    // stage Q tile (async, cooperative): this block's rows, q slice (col base h*128)
    {
        const u16* gq = qkv + (size_t)(b * 2048 + qbase + srow) * 6144 + h * 128 + shalf * 64;
        u16* dq = &Qs[srow * LDQ + shalf * 64];
#pragma unroll
        for (int j = 0; j < 8; ++j) async_b128(dq + j * 8, gq + j * 8);
    }

    auto stage_kv = [&](int buf, int kt) {
        size_t grow = (size_t)(b * 2048 + kt * 64 + srow) * 6144;
        const u16* gk = qkv + grow + 2048 + h * 128 + shalf * 64;
        const u16* gv = qkv + grow + 4096 + h * 128 + shalf * 64;
        u16* dk = &Ks[buf][srow * LDQ + shalf * 64];
        u16* dv = &Vs[buf][srow * LDQ + shalf * 64];
#pragma unroll
        for (int j = 0; j < 8; ++j) async_b128(dk + j * 8, gk + j * 8);
#pragma unroll
        for (int j = 0; j < 8; ++j) async_b128(dv + j * 8, gv + j * 8);
    };

    float m_i[8], l_i[8];
    v8f acc_o[8];
#pragma unroll
    for (int r = 0; r < 8; ++r) { m_i[r] = -3.0e38f; l_i[r] = 0.f; }
#pragma unroll
    for (int nf = 0; nf < 8; ++nf) acc_o[nf] = zero8();

    const int g = lane >> 4, n = lane & 15;

    stage_kv(0, 0);
    wait_async0();
    __syncthreads();

    int buf = 0;
    for (int kt = 0; kt < 32; ++kt) {
        if (kt + 1 < 32) stage_kv(buf ^ 1, kt + 1);   // prefetch next kv tile

        // S(16x64) = Q(16x128) * K^T, wave handles rows 16*wave..16*wave+15
        v8f s[4] = {zero8(), zero8(), zero8(), zero8()};
#pragma unroll
        for (int ks = 0; ks < 4; ++ks) {
            v16bf a = load_frag_a(Qs, wave * 16, LDQ, ks * 32);
            v16bf bb[4];
#pragma unroll
            for (int nf = 0; nf < 4; ++nf)
                bb[nf] = load_frag_b_nt(Ks[buf], nf * 16, LDQ, ks * 32);
#pragma unroll
            for (int nf = 0; nf < 4; ++nf)
                s[nf] = wmma_bf16(a, bb[nf], s[nf]);
        }

        // online softmax over the 64 keys; row of slot r is (g*8 + r)
#pragma unroll
        for (int r = 0; r < 8; ++r) {
            float mx = -3.0e38f;
#pragma unroll
            for (int nf = 0; nf < 4; ++nf) {
                s[nf][r] = s[nf][r] * scale;
                mx = fmaxf(mx, s[nf][r]);
            }
#pragma unroll
            for (int off = 1; off < 16; off <<= 1)
                mx = fmaxf(mx, __shfl_xor(mx, off, 16));
            float mnew = fmaxf(m_i[r], mx);
            float alpha = __expf(m_i[r] - mnew);
            float rsum = 0.f;
#pragma unroll
            for (int nf = 0; nf < 4; ++nf) {
                float p = __expf(s[nf][r] - mnew);
                rsum += p;
                Ps[(wave * 16 + g * 8 + r) * LDP + nf * 16 + n] = f2bf(p);
            }
#pragma unroll
            for (int off = 1; off < 16; off <<= 1)
                rsum += __shfl_xor(rsum, off, 16);
            l_i[r] = l_i[r] * alpha + rsum;
            m_i[r] = mnew;
#pragma unroll
            for (int nf = 0; nf < 8; ++nf) acc_o[nf][r] = acc_o[nf][r] * alpha;
        }

        // O += P(16x64) * V(64x128); P rows are wave-private -> no barrier needed
#pragma unroll
        for (int ks = 0; ks < 2; ++ks) {
            v16bf a = load_frag_a(Ps, wave * 16, LDP, ks * 32);
#pragma unroll
            for (int hgrp = 0; hgrp < 2; ++hgrp) {
                v16bf bb[4];
#pragma unroll
                for (int j = 0; j < 4; ++j)
                    bb[j] = load_frag_b_kn(Vs[buf], (hgrp * 4 + j) * 16, LDQ, ks * 32);
#pragma unroll
                for (int j = 0; j < 4; ++j)
                    acc_o[hgrp * 4 + j] = wmma_bf16(a, bb[j], acc_o[hgrp * 4 + j]);
            }
        }

        wait_async0();      // prefetched kv tile landed
        __syncthreads();
        buf ^= 1;
    }

    // normalize and write y in [B,T,H*D] layout (== transpose(0,2,1,3) flatten)
#pragma unroll
    for (int nf = 0; nf < 8; ++nf)
#pragma unroll
        for (int r = 0; r < 8; ++r) {
            int t = qbase + wave * 16 + g * 8 + r;
            float val = acc_o[nf][r] / l_i[r];
            y[(size_t)(b * 2048 + t) * 2048 + h * 128 + nf * 16 + n] = f2bf(val);
        }
}

// ---------- host launcher ----------
extern "C" void kernel_launch(void* const* d_in, const int* in_sizes, int n_in,
                              void* d_out, int out_size, void* d_ws, size_t ws_size,
                              hipStream_t stream) {
    (void)in_sizes; (void)n_in; (void)out_size; (void)ws_size;
    const float* x      = (const float*)d_in[0];   // [4,2048,2048]
    const float* w_attn = (const float*)d_in[1];   // [6144,2048]
    const float* w_proj = (const float*)d_in[2];   // [2048,2048]
    float* out = (float*)d_out;                    // [4,2048,2048]

    char* ws = (char*)d_ws;
    size_t off = 0;
    auto carve = [&](size_t elems) {
        u16* p = (u16*)(ws + off);
        off += ((elems * sizeof(u16) + 255) & ~(size_t)255);
        return p;
    };
    u16* xb  = carve(8192ull * 2048);   //  33.5 MB
    u16* wab = carve(6144ull * 2048);   //  25.2 MB
    u16* wpb = carve(2048ull * 2048);   //   8.4 MB
    u16* qkv = carve(8192ull * 6144);   // 100.7 MB
    u16* yb  = carve(8192ull * 2048);   //  33.5 MB

    cast_bf16_kernel<<<16777216 / 256, 256, 0, stream>>>(x, xb, 16777216);
    cast_bf16_kernel<<<12582912 / 256, 256, 0, stream>>>(w_attn, wab, 12582912);
    cast_bf16_kernel<<<4194304 / 256, 256, 0, stream>>>(w_proj, wpb, 4194304);

    // qkv[8192,6144] = x[8192,2048] @ w_attn[6144,2048]^T
    gemm_bf16_nt<u16><<<dim3(6144 / 128, 8192 / 128), 256, 0, stream>>>(
        xb, wab, qkv, 8192, 6144, 2048);

    rope_kernel<<<16777216 / 256, 256, 0, stream>>>(qkv);

    flash_attn_kernel<<<dim3(32, 16, 4), 128, 0, stream>>>(qkv, yb);

    // out[8192,2048] = y[8192,2048] @ w_proj[2048,2048]^T  (f32 out)
    gemm_bf16_nt<float><<<dim3(2048 / 128, 8192 / 128), 256, 0, stream>>>(
        yb, wpb, out, 8192, 2048, 2048);
}
